// GAT_14370960572522
// MI455X (gfx1250) — matrix operand order repaired and checked by reading, not compiled, not run on hardware
//
#include <hip/hip_runtime.h>
#include <math.h>

typedef __attribute__((ext_vector_type(16))) _Float16 v16h;
typedef __attribute__((ext_vector_type(8)))  _Float16 v8h;
typedef __attribute__((ext_vector_type(8)))  float    v8f;

#define NEG_SLOPE 0.2f
#define BN_EPS    1e-5f

// ---------------- atomic float max (sign-split trick, init to -inf) ----------
__device__ __forceinline__ void atomicMaxF(float* addr, float v) {
  if (v >= 0.f) atomicMax((int*)addr, __float_as_int(v));
  else          atomicMin((unsigned int*)addr, __float_as_uint(v));
}

// ---------------- pack / fill ----------------
__global__ void k_f32_to_f16(const float* __restrict__ s, _Float16* __restrict__ d, int n) {
  int i = blockIdx.x * blockDim.x + threadIdx.x;
  if (i < n) d[i] = (_Float16)s[i];
}

// Wt[n*K + k] = W[k*Ncols + n]  (zero-pad columns up to NcolsPad)
__global__ void k_pack_wt(const float* __restrict__ W, _Float16* __restrict__ Wt,
                          int K, int Ncols, int NcolsPad) {
  int i = blockIdx.x * blockDim.x + threadIdx.x;
  if (i >= NcolsPad * K) return;
  int n = i / K, k = i - n * K;
  Wt[i] = (n < Ncols) ? (_Float16)W[k * Ncols + n] : (_Float16)0.0f;
}

__global__ void k_fill(float* __restrict__ p, float v, int n) {
  int i = blockIdx.x * blockDim.x + threadIdx.x;
  if (i < n) p[i] = v;
}

// ---------------- WMMA GEMM strip kernel ------------------------------------
// C[M,LDC] = A[M,128] x B[128,STOREN];  A: f16 row-major, Bt: f16 [NTILES*16,128].
// One wave per 16-row strip; A K-fragments register-resident across all column
// tiles. LDC/STOREN are compile-time so store addressing is immediate-offset.
template<int NTILES, int STOREN, int LDC>
__global__ void __launch_bounds__(32)
k_wmma_gemm_strip(const _Float16* __restrict__ A, const _Float16* __restrict__ Bt,
                  float* __restrict__ C, int Mrows) {
  const int K = 128;
  int lane = threadIdx.x & 31;
  int l16  = lane & 15;
  int hi   = lane >> 4;                 // half-wave -> K sub-stripe
  int tm   = blockIdx.x;
  int m    = tm * 16 + l16;
  int mA   = m < Mrows ? m : (Mrows - 1);
  const _Float16* arow = A + (size_t)mA * K;

  // Load the 4 A fragments (16x32 f16 each) once; keep in registers.
  v16h afrag[4];
#pragma unroll
  for (int kc4 = 0; kc4 < 4; ++kc4) {
    v8h alo = *(const v8h*)(arow + kc4 * 32 + hi * 8);
    v8h ahi = *(const v8h*)(arow + kc4 * 32 + 16 + hi * 8);
#pragma unroll
    for (int i = 0; i < 8; ++i) { afrag[kc4][i] = alo[i]; afrag[kc4][i + 8] = ahi[i]; }
  }

  const bool fullM = (tm * 16 + 16) <= Mrows;

#pragma unroll
  for (int tn = 0; tn < NTILES; ++tn) {
    const _Float16* brow = Bt + (size_t)(tn * 16 + l16) * K;
    v8f acc = {};
#pragma unroll
    for (int kc4 = 0; kc4 < 4; ++kc4) {
      v8h blo = *(const v8h*)(brow + kc4 * 32 + hi * 8);
      v8h bhi = *(const v8h*)(brow + kc4 * 32 + 16 + hi * 8);
      v16h b;
#pragma unroll
      for (int i = 0; i < 8; ++i) { b[i] = blo[i]; b[i + 8] = bhi[i]; }
      acc = __builtin_amdgcn_wmma_f32_16x16x32_f16(false, afrag[kc4], false, b,
                                                   (short)0, acc, false, false);
    }
    // C layout: VGPR r -> row (r + 8*hi), col = l16
    int col = tn * 16 + l16;
    // compile-time fold: all tiles except a ragged last one skip the predicate
    bool colOK = ((tn + 1) * 16 <= STOREN) ? true : (col < STOREN);
    if (colOK) {
      float* cp = C + (size_t)(tm * 16 + 8 * hi) * LDC + col;
      if (fullM) {
#pragma unroll
        for (int r = 0; r < 8; ++r) cp[r * LDC] = acc[r];   // immediate offsets
      } else {
#pragma unroll
        for (int r = 0; r < 8; ++r)
          if (tm * 16 + 8 * hi + r < Mrows) cp[r * LDC] = acc[r];
      }
    }
  }
}

// ---------------- per-node attention terms ----------------
template<int H, int HID, int LDF>
__global__ void k_node_alpha(const float* __restrict__ feat,
                             const float* __restrict__ a_src, const float* __restrict__ a_dst,
                             float* __restrict__ als, float* __restrict__ ald, int Nn) {
  int t = blockIdx.x * blockDim.x + threadIdx.x;
  if (t >= Nn * H) return;
  int node = t / H, h = t - node * H;
  const float* f = feat + (size_t)node * LDF + h * HID;
  float ss = 0.f, sd = 0.f;
#pragma unroll 4
  for (int c = 0; c < HID; ++c) {
    float v = f[c];
    ss += v * a_src[h * HID + c];
    sd += v * a_dst[h * HID + c];
  }
  als[t] = ss; ald[t] = sd;
}

// ---------------- edge passes ----------------
template<int H>
__global__ void k_edge_logit_max(const int* __restrict__ ei, int E, int ET,
                                 const float* __restrict__ als, const float* __restrict__ ald,
                                 float* __restrict__ logits, float* __restrict__ mbuf) {
  int t = blockIdx.x * blockDim.x + threadIdx.x;
  if (t >= ET * H) return;
  int e = t / H, h = t - e * H;
  int s = e < E ? ei[e]     : (e - E);   // self loop past E
  int d = e < E ? ei[E + e] : (e - E);
  float v = als[s * H + h] + ald[d * H + h];
  v = v > 0.f ? v : NEG_SLOPE * v;       // LeakyReLU
  logits[t] = v;
  atomicMaxF(mbuf + d * H + h, v);
}

template<int H>
__global__ void k_edge_expsum(const int* __restrict__ ei, int E, int ET,
                              float* __restrict__ logits,
                              const float* __restrict__ mbuf, float* __restrict__ sbuf) {
  int t = blockIdx.x * blockDim.x + threadIdx.x;
  if (t >= ET * H) return;
  int e = t / H, h = t - e * H;
  int d = e < E ? ei[E + e] : (e - E);
  float v = expf(logits[t] - mbuf[d * H + h]);
  logits[t] = v;                          // overwrite with exp value
  atomicAdd(sbuf + d * H + h, v);
}

template<int C, int HID, int H>
__global__ void k_edge_aggregate(const int* __restrict__ ei, int E, int ET,
                                 const float* __restrict__ feat,
                                 const float* __restrict__ ebuf, const float* __restrict__ sbuf,
                                 float* __restrict__ agg) {
  long long t = (long long)blockIdx.x * blockDim.x + threadIdx.x;
  if (t >= (long long)ET * C) return;
  int e = (int)(t / C);
  int c = (int)(t - (long long)e * C);
  int s = e < E ? ei[e]     : (e - E);
  int d = e < E ? ei[E + e] : (e - E);
  int h = c / HID;
  float alpha = ebuf[e * H + h] / (sbuf[d * H + h] + 1e-16f);
  atomicAdd(agg + (size_t)d * C + c, feat[(size_t)s * C + c] * alpha);
}

// ---------------- layer-1 epilogue: +bias, BN(eval), ReLU, -> f16 ----------
__global__ void k_bn_relu_to_f16(const float* __restrict__ agg, const float* __restrict__ bias,
                                 const float* __restrict__ gamma, const float* __restrict__ beta,
                                 const float* __restrict__ mean, const float* __restrict__ var,
                                 _Float16* __restrict__ out, int total, int C) {
  int t = blockIdx.x * blockDim.x + threadIdx.x;
  if (t >= total) return;
  int c = t % C;
  float v = agg[t] + bias[c];
  v = (v - mean[c]) * rsqrtf(var[c] + BN_EPS) * gamma[c] + beta[c];
  out[t] = (_Float16)(v > 0.f ? v : 0.f);
}

// ---------------- output epilogue: +bias, log_softmax over 40 classes -------
__global__ void k_bias_logsoftmax40(const float* __restrict__ agg, const float* __restrict__ b,
                                    float* __restrict__ out, int Nn) {
  int i = blockIdx.x * blockDim.x + threadIdx.x;
  if (i >= Nn) return;
  float v[40];
  float mx = -INFINITY;
#pragma unroll
  for (int c = 0; c < 40; ++c) { v[c] = agg[i * 40 + c] + b[c]; mx = fmaxf(mx, v[c]); }
  float s = 0.f;
#pragma unroll
  for (int c = 0; c < 40; ++c) s += expf(v[c] - mx);
  float ls = logf(s);
#pragma unroll
  for (int c = 0; c < 40; ++c) out[i * 40 + c] = v[c] - mx - ls;
}

// ------------------------------------------------------------------
static inline int cdiv(long long a, int b) { return (int)((a + b - 1) / b); }

extern "C" void kernel_launch(void* const* d_in, const int* in_sizes, int n_in,
                              void* d_out, int out_size, void* d_ws, size_t ws_size,
                              hipStream_t stream) {
  (void)n_in; (void)out_size; (void)ws_size;
  const float* x    = (const float*)d_in[0];
  const int*   ei   = (const int*)  d_in[1];
  const float* W1   = (const float*)d_in[2];
  const float* a1s  = (const float*)d_in[3];
  const float* a1d  = (const float*)d_in[4];
  const float* b1   = (const float*)d_in[5];
  const float* gam1 = (const float*)d_in[6];
  const float* bet1 = (const float*)d_in[7];
  const float* mu1  = (const float*)d_in[8];
  const float* var1 = (const float*)d_in[9];
  const float* W2   = (const float*)d_in[10];
  const float* a2s  = (const float*)d_in[11];
  const float* a2d  = (const float*)d_in[12];
  const float* b2   = (const float*)d_in[13];

  const int Nn = in_sizes[0] / 128;   // 50000
  const int E  = in_sizes[1] / 2;     // 800000
  const int ET = E + Nn;              // + self loops

  // -------- workspace carving (layer-2 temps alias dead layer-1 temps) -----
  char* base = (char*)d_ws;
  size_t off = 0;
  auto take = [&](size_t bytes) -> void* {
    off = (off + 255) & ~(size_t)255;
    void* p = base + off;
    off += bytes;
    return p;
  };
  _Float16* xh  = (_Float16*)take((size_t)Nn * 128 * 2);
  _Float16* w1t = (_Float16*)take(128 * 128 * 2);
  _Float16* w2t = (_Float16*)take(48 * 128 * 2);
  _Float16* h2h = (_Float16*)take((size_t)Nn * 128 * 2);
  size_t mark = off;
  float* h1   = (float*)take((size_t)Nn * 128 * 4);
  float* agg1 = (float*)take((size_t)Nn * 128 * 4);
  float* als1 = (float*)take((size_t)Nn * 2 * 4);
  float* ald1 = (float*)take((size_t)Nn * 2 * 4);
  float* m1   = (float*)take((size_t)Nn * 2 * 4);
  float* s1   = (float*)take((size_t)Nn * 2 * 4);
  float* lg1  = (float*)take((size_t)ET * 2 * 4);
  off = mark;  // layer-1 temps dead before layer-2 temps are written
  float* h3   = (float*)take((size_t)Nn * 40 * 4);
  float* agg2 = (float*)take((size_t)Nn * 40 * 4);
  float* als2 = (float*)take((size_t)Nn * 4);
  float* ald2 = (float*)take((size_t)Nn * 4);
  float* m2   = (float*)take((size_t)Nn * 4);
  float* s2   = (float*)take((size_t)Nn * 4);
  float* lg2  = (float*)take((size_t)ET * 4);

  const int B = 256;

  // -------- pack operands to f16 (A row-major, B transposed) --------
  k_f32_to_f16<<<cdiv((long long)Nn * 128, B), B, 0, stream>>>(x, xh, Nn * 128);
  k_pack_wt<<<cdiv(128 * 128, B), B, 0, stream>>>(W1, w1t, 128, 128, 128);
  k_pack_wt<<<cdiv(48 * 128, B), B, 0, stream>>>(W2, w2t, 128, 40, 48);

  // -------- layer 1: h1 = x @ W1  (WMMA, one wave per 16-row strip) --------
  k_wmma_gemm_strip<8, 128, 128><<<(Nn + 15) / 16, 32, 0, stream>>>(xh, w1t, h1, Nn);

  // attention terms
  k_node_alpha<2, 64, 128><<<cdiv((long long)Nn * 2, B), B, 0, stream>>>(h1, a1s, a1d, als1, ald1, Nn);

  // init segment buffers
  k_fill<<<cdiv((long long)Nn * 2, B), B, 0, stream>>>(m1, -INFINITY, Nn * 2);
  k_fill<<<cdiv((long long)Nn * 2, B), B, 0, stream>>>(s1, 0.f, Nn * 2);
  k_fill<<<cdiv((long long)Nn * 128, B), B, 0, stream>>>(agg1, 0.f, Nn * 128);

  // segment softmax + aggregation
  k_edge_logit_max<2><<<cdiv((long long)ET * 2, B), B, 0, stream>>>(ei, E, ET, als1, ald1, lg1, m1);
  k_edge_expsum<2><<<cdiv((long long)ET * 2, B), B, 0, stream>>>(ei, E, ET, lg1, m1, s1);
  k_edge_aggregate<128, 64, 2><<<cdiv((long long)ET * 128, B), B, 0, stream>>>(ei, E, ET, h1, lg1, s1, agg1);

  // epilogue: bias + BN + ReLU -> f16 activations for layer 2
  k_bn_relu_to_f16<<<cdiv((long long)Nn * 128, B), B, 0, stream>>>(agg1, b1, gam1, bet1, mu1, var1,
                                                                   h2h, Nn * 128, 128);

  // -------- layer 2: h3 = h2 @ W2  (WMMA, N padded 40->48) --------
  k_wmma_gemm_strip<3, 40, 40><<<(Nn + 15) / 16, 32, 0, stream>>>(h2h, w2t, h3, Nn);

  k_node_alpha<1, 40, 40><<<cdiv(Nn, B), B, 0, stream>>>(h3, a2s, a2d, als2, ald2, Nn);

  k_fill<<<cdiv(Nn, B), B, 0, stream>>>(m2, -INFINITY, Nn);
  k_fill<<<cdiv(Nn, B), B, 0, stream>>>(s2, 0.f, Nn);
  k_fill<<<cdiv((long long)Nn * 40, B), B, 0, stream>>>(agg2, 0.f, Nn * 40);

  k_edge_logit_max<1><<<cdiv(ET, B), B, 0, stream>>>(ei, E, ET, als2, ald2, lg2, m2);
  k_edge_expsum<1><<<cdiv(ET, B), B, 0, stream>>>(ei, E, ET, lg2, m2, s2);
  k_edge_aggregate<40, 40, 1><<<cdiv((long long)ET * 40, B), B, 0, stream>>>(ei, E, ET, h3, lg2, s2, agg2);

  // output: +bias, log_softmax
  k_bias_logsoftmax40<<<cdiv(Nn, B), B, 0, stream>>>(agg2, b2, (float*)d_out, Nn);
}
